// TransformerEncoderLayer_35373350649949
// MI455X (gfx1250) — matrix operand order repaired and checked by reading, not compiled, not run on hardware
//
#include <hip/hip_runtime.h>

// ---------------------------------------------------------------------------
// Transformer encoder layer, MI455X (gfx1250, wave32, WMMA bf16 16x16x32)
//   B=2, S=2048, D=1024, H=16, DK=64, DFF=4096
// ---------------------------------------------------------------------------

#define D_MODEL 1024
#define N_HEADS 16
#define D_K     64
#define D_FF    4096
#define SEQ     2048
#define BATCH   2
#define ROWS    (BATCH * SEQ)   // 4096

typedef __attribute__((ext_vector_type(16))) __bf16 v16bf;
typedef __attribute__((ext_vector_type(8)))  float  v8f;
typedef __attribute__((ext_vector_type(4)))  int    v4i;

union FragB {
    v16bf v;
    uint4 q[2];
    unsigned short h[16];
};

__device__ __forceinline__ unsigned short f32_to_bf16(float f) {
    union { float f; unsigned int u; } c; c.f = f;
    unsigned int u = c.u;
    unsigned int r = u + 0x7FFFu + ((u >> 16) & 1u);   // round-to-nearest-even
    return (unsigned short)(r >> 16);
}

// ---------------------------------------------------------------------------
// Async global -> LDS staging (CDNA5 GLOBAL_LOAD_ASYNC_TO_LDS_B128, ASYNCcnt)
// with a synchronous fallback if the builtin is unavailable.
// Builtin signature (from compiler diagnostic): (v4i AS1* src, v4i AS3* dst,
// imm offset, imm cpol).
// ---------------------------------------------------------------------------
#if defined(__has_builtin)
#  if __has_builtin(__builtin_amdgcn_global_load_async_to_lds_b128)
#    define HAVE_ASYNC_LDS 1
#  endif
#endif
#ifndef HAVE_ASYNC_LDS
#  define HAVE_ASYNC_LDS 0
#endif

#define AS1 __attribute__((address_space(1)))
#define AS3 __attribute__((address_space(3)))

__device__ __forceinline__ void copy16_to_lds(const unsigned short* g,
                                              unsigned short* l)
{
#if HAVE_ASYNC_LDS
    __builtin_amdgcn_global_load_async_to_lds_b128(
        (AS1 v4i*)const_cast<unsigned short*>(g), (AS3 v4i*)l, 0, 0);
#else
    *(uint4*)l = *(const uint4*)g;
#endif
}

__device__ __forceinline__ void wait_async_lds()
{
#if HAVE_ASYNC_LDS
#  if __has_builtin(__builtin_amdgcn_s_wait_asynccnt)
    __builtin_amdgcn_s_wait_asynccnt(0);
#  else
    asm volatile("s_wait_asynccnt 0x0" ::: "memory");
#  endif
#endif
}

// ---------------------------------------------------------------------------
// fp32 [K][N] weight  ->  bf16 transposed [N][K]
// ---------------------------------------------------------------------------
__global__ __launch_bounds__(256) void convT_kernel(
    const float* __restrict__ W, unsigned short* __restrict__ Wt, int K, int N)
{
    size_t i = (size_t)blockIdx.x * 256 + threadIdx.x;
    if (i >= (size_t)K * N) return;
    int n = (int)(i % N);
    int k = (int)(i / N);
    Wt[(size_t)n * K + k] = f32_to_bf16(W[i]);
}

// ---------------------------------------------------------------------------
// LayerNorm: fp32 row -> bf16 row (D=1024, 256 threads/row)
// ---------------------------------------------------------------------------
__global__ __launch_bounds__(256) void ln_kernel(
    const float* __restrict__ X, const float* __restrict__ g,
    const float* __restrict__ bt, unsigned short* __restrict__ out)
{
    __shared__ float red[2][8];
    const int row  = blockIdx.x;
    const int tid  = threadIdx.x;
    const int wave = tid >> 5, lane = tid & 31;
    const float* xr = X + (size_t)row * D_MODEL;
    float4 xv = *(const float4*)&xr[tid * 4];
    float s  = xv.x + xv.y + xv.z + xv.w;
    float s2 = xv.x * xv.x + xv.y * xv.y + xv.z * xv.z + xv.w * xv.w;
    for (int m = 1; m < 32; m <<= 1) {
        s  += __shfl_xor(s,  m, 32);
        s2 += __shfl_xor(s2, m, 32);
    }
    if (lane == 0) { red[0][wave] = s; red[1][wave] = s2; }
    __syncthreads();
    if (wave == 0) {
        float a = (lane < 8) ? red[0][lane] : 0.f;
        float b = (lane < 8) ? red[1][lane] : 0.f;
        for (int m = 1; m < 8; m <<= 1) {
            a += __shfl_xor(a, m, 32);
            b += __shfl_xor(b, m, 32);
        }
        if (lane == 0) { red[0][0] = a; red[1][0] = b; }
    }
    __syncthreads();
    const float mean = red[0][0] * (1.f / D_MODEL);
    const float var  = red[1][0] * (1.f / D_MODEL) - mean * mean;
    const float rstd = rsqrtf(var + 1e-5f);
    float xe[4] = { xv.x, xv.y, xv.z, xv.w };
    for (int i = 0; i < 4; ++i) {
        int c = tid * 4 + i;
        out[(size_t)row * D_MODEL + c] = f32_to_bf16((xe[i] - mean) * rstd * g[c] + bt[c]);
    }
}

// ---------------------------------------------------------------------------
// Transpose V [B*S][D] bf16 -> Vt [B*H][DK][S] bf16
// ---------------------------------------------------------------------------
__global__ __launch_bounds__(256) void vtrans_kernel(
    const unsigned short* __restrict__ V, unsigned short* __restrict__ Vt)
{
    size_t i = (size_t)blockIdx.x * 256 + threadIdx.x;
    const size_t total = (size_t)BATCH * N_HEADS * D_K * SEQ;
    if (i >= total) return;
    int s = (int)(i % SEQ); size_t r = i / SEQ;
    int d = (int)(r % D_K); r /= D_K;
    int h = (int)(r % N_HEADS);
    int b = (int)(r / N_HEADS);
    Vt[i] = V[((size_t)(b * SEQ + s)) * D_MODEL + h * D_K + d];
}

// ---------------------------------------------------------------------------
// GEMM: C[M][N] = act(A[M][K] @ Bt[N][K]^T + bias)  (+ residual)
//   Block tile 128x128, 8 waves, wave tile 32x64 (2x4 accumulators),
//   K-step 32, double-buffered LDS with async-to-LDS prefetch pipeline.
// ---------------------------------------------------------------------------
template <int RELU, int RESID>
__global__ __launch_bounds__(256) void gemm_bf16(
    const unsigned short* __restrict__ A,   // [M][K]
    const unsigned short* __restrict__ Bt,  // [N][K]
    const float* __restrict__ bias,         // [N]
    const float* __restrict__ resid,        // [M][N] (RESID)
    float* __restrict__ outF,               // [M][N] (RESID)
    unsigned short* __restrict__ outB,      // [M][N] (!RESID)
    int M, int N, int K)
{
    __shared__ unsigned short lA[2][128 * 32];   // 16 KB
    __shared__ unsigned short lB[2][128 * 32];   // 16 KB
    const int tid  = threadIdx.x;
    const int wave = tid >> 5, lane = tid & 31;
    const int wm = wave & 3, wn = wave >> 2;     // 4 x 2 waves
    const int bm = blockIdx.y * 128, bn = blockIdx.x * 128;
    const int half = lane >> 4;    // 0: lanes 0-15, 1: lanes 16-31
    const int l16  = lane & 15;

    auto issue_tile = [&](int buf, int k0) {
        // A tile 128x32 and Bt tile 128x32: 512 b128 chunks each, 2+2/thread
        for (int c = tid; c < 512; c += 256) {
            int row = c >> 2, qi = c & 3;
            copy16_to_lds(&A [(size_t)(bm + row) * K + k0 + qi * 8],
                          &lA[buf][row * 32 + qi * 8]);
            copy16_to_lds(&Bt[(size_t)(bn + row) * K + k0 + qi * 8],
                          &lB[buf][row * 32 + qi * 8]);
        }
    };

    const int KT = K / 32;
    v8f acc[2][4] = {};
    issue_tile(0, 0);
    for (int kt = 0; kt < KT; ++kt) {
        wait_async_lds();
        __syncthreads();
        if (kt + 1 < KT) issue_tile((kt + 1) & 1, (kt + 1) * 32);

        const unsigned short* cA = lA[kt & 1];
        const unsigned short* cB = lB[kt & 1];
        FragB a[2], b[4];
        for (int mi = 0; mi < 2; ++mi) {
            int row = wm * 32 + mi * 16 + l16;
            // A layout: lanes<16 K=0..7 & 16..23 ; lanes>=16 K=8..15 & 24..31
            a[mi].q[0] = *(const uint4*)&cA[row * 32 + half * 8];
            a[mi].q[1] = *(const uint4*)&cA[row * 32 + 16 + half * 8];
        }
        for (int ni = 0; ni < 4; ++ni) {
            int row = wn * 64 + ni * 16 + l16;
            // B layout: lanes<16 K=0..7 & 16..23 ; lanes>=16 K=8..15 & 24..31
            b[ni].q[0] = *(const uint4*)&cB[row * 32 + half * 8];
            b[ni].q[1] = *(const uint4*)&cB[row * 32 + 16 + half * 8];
        }
        for (int mi = 0; mi < 2; ++mi)
            for (int ni = 0; ni < 4; ++ni)
                acc[mi][ni] = __builtin_amdgcn_wmma_f32_16x16x32_bf16(
                    false, a[mi].v, false, b[ni].v,
                    (short)0, acc[mi][ni], false, false);
    }

    // epilogue: C layout lane=col (N=l16), VGPR r = row r + 8*half
    for (int mi = 0; mi < 2; ++mi)
        for (int ni = 0; ni < 4; ++ni) {
            int col = bn + wn * 64 + ni * 16 + l16;
            float bv = bias[col];
            for (int r = 0; r < 8; ++r) {
                int row = bm + wm * 32 + mi * 16 + r + half * 8;
                float v = acc[mi][ni][r] + bv;
                if (RELU) v = v > 0.f ? v : 0.f;
                if (RESID)
                    outF[(size_t)row * N + col] = resid[(size_t)row * N + col] + v;
                else
                    outB[(size_t)row * N + col] = f32_to_bf16(v);
            }
        }
}

// ---------------------------------------------------------------------------
// Flash attention (per head). Computes S^T = K·Q^T so the softmaxed P tile
// lands directly in the WMMA A-fragment layout for P·V (no lane shuffles).
//   grid = (S/64, B*H), 128 threads (4 waves), wave = 16 query rows.
// ---------------------------------------------------------------------------
__global__ __launch_bounds__(128) void attn_kernel(
    const unsigned short* __restrict__ Q,   // [B*S][D] bf16
    const unsigned short* __restrict__ Kb,  // [B*S][D] bf16
    const unsigned short* __restrict__ Vt,  // [B*H][DK][S] bf16
    unsigned short* __restrict__ Ctx)       // [B*S][D] bf16
{
    __shared__ unsigned short lK[32 * 64];  // K tile [t][d]
    __shared__ unsigned short lV[64 * 32];  // V^T tile [d][t]
    const int bh = blockIdx.y;
    const int b = bh >> 4, h = bh & 15;
    const int tid  = threadIdx.x;
    const int wave = tid >> 5, lane = tid & 31;
    const int half = lane >> 4, l16 = lane & 15;
    const int q0 = blockIdx.x * 64 + wave * 16;

    const unsigned short* qp = Q  + (size_t)(b * SEQ) * D_MODEL + h * D_K;
    const unsigned short* kp = Kb + (size_t)(b * SEQ) * D_MODEL + h * D_K;
    const unsigned short* vp = Vt + (size_t)bh * D_K * SEQ;

    // Q fragments (B-side of S^T wmma), one per d-half, resident all loop
    FragB qf[2];
    {
        const unsigned short* base = qp + (size_t)(q0 + l16) * D_MODEL;
        for (int c = 0; c < 2; ++c) {
            int d0 = c * 32 + half * 8;
            qf[c].q[0] = *(const uint4*)(base + d0);
            qf[c].q[1] = *(const uint4*)(base + d0 + 16);
        }
    }

    float m_run = -1e30f, l_run = 0.f;
    v8f acc[4] = {};   // ctx 16q x 64d as 4 tiles along d

    for (int t0 = 0; t0 < SEQ; t0 += 32) {
        __syncthreads();
        for (int c = tid; c < 256; c += 128) {          // K tile 32x64
            int row = c >> 3, qi = c & 7;
            copy16_to_lds(&kp[(size_t)(t0 + row) * D_MODEL + qi * 8],
                          &lK[row * 64 + qi * 8]);
        }
        for (int c = tid; c < 256; c += 128) {          // V^T tile 64x32
            int row = c >> 2, qi = c & 3;
            copy16_to_lds(&vp[(size_t)row * SEQ + t0 + qi * 8],
                          &lV[row * 32 + qi * 8]);
        }
        wait_async_lds();
        __syncthreads();

        // S^T tiles: M=t (two 16-row tiles), N=q, K-dim = d (64 = 2x32)
        v8f st[2];
        for (int tt = 0; tt < 2; ++tt) {
            int row = tt * 16 + l16;
            FragB a0, a1;
            a0.q[0] = *(const uint4*)&lK[row * 64 + half * 8];
            a0.q[1] = *(const uint4*)&lK[row * 64 + 16 + half * 8];
            a1.q[0] = *(const uint4*)&lK[row * 64 + 32 + half * 8];
            a1.q[1] = *(const uint4*)&lK[row * 64 + 48 + half * 8];
            v8f z = {};
            z = __builtin_amdgcn_wmma_f32_16x16x32_bf16(
                    false, a0.v, false, qf[0].v, (short)0, z, false, false);
            z = __builtin_amdgcn_wmma_f32_16x16x32_bf16(
                    false, a1.v, false, qf[1].v, (short)0, z, false, false);
            st[tt] = z;
        }

        // online softmax; lane n holds query q = n%16 stats
        const float sc = 0.125f;   // 1/sqrt(64)
        float pm = -1e30f;
        for (int tt = 0; tt < 2; ++tt)
            for (int r = 0; r < 8; ++r) {
                float v = st[tt][r] * sc;
                st[tt][r] = v;
                pm = fmaxf(pm, v);
            }
        pm = fmaxf(pm, __shfl_xor(pm, 16, 32));
        float m_new = fmaxf(m_run, pm);

        FragB pf;
        float psum = 0.f;
        for (int r = 0; r < 8; ++r) {
            float p0 = __expf(st[0][r] - m_new);
            float p1 = __expf(st[1][r] - m_new);
            psum += p0 + p1;
            pf.h[r]     = f32_to_bf16(p0);   // t = r  + 8*half   (A-frag K 0..7)
            pf.h[8 + r] = f32_to_bf16(p1);   // t = 16 + r + 8*half (K 16..23)
        }
        psum += __shfl_xor(psum, 16, 32);
        float scale = __expf(m_run - m_new);
        l_run = l_run * scale + psum;
        m_run = m_new;

        // redistribute per-query scale to ctx accumulator layout
        float s8[8];
        for (int r = 0; r < 8; ++r)
            s8[r] = __shfl(scale, r + half * 8, 32);
        for (int dc = 0; dc < 4; ++dc)
            for (int r = 0; r < 8; ++r)
                acc[dc][r] *= s8[r];

        // ctx += P (16x32) @ V (32x64): B frags from V^T tile
        for (int dc = 0; dc < 4; ++dc) {
            int row = dc * 16 + l16;
            FragB vf;
            vf.q[0] = *(const uint4*)&lV[row * 32 + half * 8];
            vf.q[1] = *(const uint4*)&lV[row * 32 + 16 + half * 8];
            acc[dc] = __builtin_amdgcn_wmma_f32_16x16x32_bf16(
                false, pf.v, false, vf.v, (short)0, acc[dc], false, false);
        }
    }

    // finalize: divide by l, store ctx bf16 [B*S][D]
    float linv = 1.f / l_run;
    float li8[8];
    for (int r = 0; r < 8; ++r)
        li8[r] = __shfl(linv, r + half * 8, 32);
    for (int dc = 0; dc < 4; ++dc)
        for (int r = 0; r < 8; ++r) {
            int row = b * SEQ + q0 + r + half * 8;
            int col = h * D_K + dc * 16 + l16;
            Ctx[(size_t)row * D_MODEL + col] = f32_to_bf16(acc[dc][r] * li8[r]);
        }
}

// ---------------------------------------------------------------------------
// launch
// ---------------------------------------------------------------------------
extern "C" void kernel_launch(void* const* d_in, const int* in_sizes, int n_in,
                              void* d_out, int out_size, void* d_ws, size_t ws_size,
                              hipStream_t stream)
{
    const float* x     = (const float*)d_in[0];
    const float* Wq    = (const float*)d_in[1];
    const float* bq    = (const float*)d_in[2];
    const float* Wk    = (const float*)d_in[3];
    const float* bk    = (const float*)d_in[4];
    const float* Wv    = (const float*)d_in[5];
    const float* bv    = (const float*)d_in[6];
    const float* Wo    = (const float*)d_in[7];
    const float* bo    = (const float*)d_in[8];
    const float* ln1_g = (const float*)d_in[9];
    const float* ln1_b = (const float*)d_in[10];
    const float* W1    = (const float*)d_in[11];
    const float* b1    = (const float*)d_in[12];
    const float* W2    = (const float*)d_in[13];
    const float* b2    = (const float*)d_in[14];
    const float* ln2_g = (const float*)d_in[15];
    const float* ln2_b = (const float*)d_in[16];

    char* ws = (char*)d_ws;
    const size_t MB = 1u << 20;
    unsigned short* hbuf = (unsigned short*)(ws + 0);        //  8 MB  [4096][1024] bf16
    unsigned short* WqT  = (unsigned short*)(ws + 8  * MB);  //  2 MB  [1024][1024]
    unsigned short* WkT  = (unsigned short*)(ws + 10 * MB);
    unsigned short* WvT  = (unsigned short*)(ws + 12 * MB);
    unsigned short* WoT  = (unsigned short*)(ws + 14 * MB);
    unsigned short* W1T  = (unsigned short*)(ws + 16 * MB);  //  8 MB  [4096][1024]
    unsigned short* W2T  = (unsigned short*)(ws + 24 * MB);  //  8 MB  [1024][4096]
    unsigned short* qbuf = (unsigned short*)(ws + 32 * MB);  //  8 MB
    unsigned short* kbuf = (unsigned short*)(ws + 40 * MB);  //  8 MB
    unsigned short* vbuf = (unsigned short*)(ws + 48 * MB);  //  8 MB
    unsigned short* vT   = (unsigned short*)(ws + 56 * MB);  //  8 MB  [32][64][2048]
    unsigned short* ctx  = (unsigned short*)(ws + 64 * MB);  //  8 MB
    float*          x1   = (float*)         (ws + 72 * MB);  // 16 MB  fp32
    unsigned short* ff1  = (unsigned short*)(ws + 88 * MB);  // 32 MB  [4096][4096]

    // weight convert + transpose (fp32 -> bf16^T)
    const int nDD = D_MODEL * D_MODEL, nDF = D_MODEL * D_FF;
    convT_kernel<<<(nDD + 255) / 256, 256, 0, stream>>>(Wq, WqT, D_MODEL, D_MODEL);
    convT_kernel<<<(nDD + 255) / 256, 256, 0, stream>>>(Wk, WkT, D_MODEL, D_MODEL);
    convT_kernel<<<(nDD + 255) / 256, 256, 0, stream>>>(Wv, WvT, D_MODEL, D_MODEL);
    convT_kernel<<<(nDD + 255) / 256, 256, 0, stream>>>(Wo, WoT, D_MODEL, D_MODEL);
    convT_kernel<<<(nDF + 255) / 256, 256, 0, stream>>>(W1, W1T, D_MODEL, D_FF);
    convT_kernel<<<(nDF + 255) / 256, 256, 0, stream>>>(W2, W2T, D_FF, D_MODEL);

    // LN1
    ln_kernel<<<ROWS, 256, 0, stream>>>(x, ln1_g, ln1_b, hbuf);

    // Q/K/V projections
    dim3 gDD(D_MODEL / 128, ROWS / 128);    // (8, 32)
    gemm_bf16<0, 0><<<gDD, 256, 0, stream>>>(hbuf, WqT, bq, nullptr, nullptr, qbuf,
                                             ROWS, D_MODEL, D_MODEL);
    gemm_bf16<0, 0><<<gDD, 256, 0, stream>>>(hbuf, WkT, bk, nullptr, nullptr, kbuf,
                                             ROWS, D_MODEL, D_MODEL);
    gemm_bf16<0, 0><<<gDD, 256, 0, stream>>>(hbuf, WvT, bv, nullptr, nullptr, vbuf,
                                             ROWS, D_MODEL, D_MODEL);
    vtrans_kernel<<<(BATCH * N_HEADS * D_K * SEQ + 255) / 256, 256, 0, stream>>>(vbuf, vT);

    // attention
    attn_kernel<<<dim3(SEQ / 64, BATCH * N_HEADS), 128, 0, stream>>>(qbuf, kbuf, vT, ctx);

    // x1 = x + ctx @ Wo + bo
    gemm_bf16<0, 1><<<gDD, 256, 0, stream>>>(ctx, WoT, bo, x, x1, nullptr,
                                             ROWS, D_MODEL, D_MODEL);

    // LN2 -> hbuf (reused)
    ln_kernel<<<ROWS, 256, 0, stream>>>(x1, ln2_g, ln2_b, hbuf);

    // ff1 = relu(h @ W1 + b1)
    dim3 gFF(D_FF / 128, ROWS / 128);       // (32, 32)
    gemm_bf16<1, 0><<<gFF, 256, 0, stream>>>(hbuf, W1T, b1, nullptr, nullptr, ff1,
                                             ROWS, D_FF, D_MODEL);

    // out = x1 + ff1 @ W2 + b2
    gemm_bf16<0, 1><<<gDD, 256, 0, stream>>>(ff1, W2T, b2, x1, (float*)d_out, nullptr,
                                             ROWS, D_MODEL, D_FF);
}
